// Attention_44994077393310
// MI455X (gfx1250) — compile-verified, matrix operand order checked by reading, not run on hardware
//
#include <hip/hip_runtime.h>

#define DIM 768
#define NHEADS 4
#define HDIM 192
#define SEQ 2048
#define BATCH 8
#define MROWS (BATCH*SEQ)                         /* 16384 */
#define WELEMS (DIM*DIM)                          /* 589824 */
#define AELEMS ((size_t)MROWS*DIM)                /* 12582912 */
#define HELEMS ((size_t)BATCH*NHEADS*SEQ*HDIM)    /* 12582912 */
// scale * log2(e) folded into Q so softmax uses exp2 directly
#define CF_SCALE (1.44269504088896f * 0.0721687836487032f)

typedef __attribute__((ext_vector_type(16))) __bf16        v16bf;
typedef __attribute__((ext_vector_type(8)))  float         v8f;
typedef __attribute__((ext_vector_type(8)))  unsigned int  v8u;
typedef __attribute__((ext_vector_type(4)))  int           v4i;

#if __has_builtin(__builtin_amdgcn_global_load_async_to_lds_b128)
#define HAVE_ASYNC 1
typedef __attribute__((address_space(1))) v4i* as1_v4i;
typedef __attribute__((address_space(3))) v4i* as3_v4i;
#endif

// 16B memory -> LDS copy; async (no VGPR round-trip, ASYNCcnt-tracked) when available
__device__ __forceinline__ void async_cp16(void* l, const void* g) {
#ifdef HAVE_ASYNC
  // flat LDS pointers carry the LDS byte offset in addr[31:0] (ISA 10.2)
  __builtin_amdgcn_global_load_async_to_lds_b128(
      (as1_v4i)(unsigned long long)(uintptr_t)g,
      (as3_v4i)(unsigned int)(uintptr_t)l, 0, 0);
#else
  *reinterpret_cast<uint4*>(l) = *reinterpret_cast<const uint4*>(g);
#endif
}

__device__ __forceinline__ void wait_async() {
#ifdef HAVE_ASYNC
#if __has_builtin(__builtin_amdgcn_s_wait_asynccnt)
  __builtin_amdgcn_s_wait_asynccnt(0);
#else
  asm volatile("s_wait_asynccnt 0x0" ::: "memory");
#endif
#endif
}

__device__ __forceinline__ unsigned short f2bf(float f) {
  unsigned u = __builtin_bit_cast(unsigned, f);
  unsigned r = u + 0x7FFFu + ((u >> 16) & 1u);   // round-to-nearest-even
  return (unsigned short)(r >> 16);
}

__device__ __forceinline__ v16bf mk_frag(uint4 lo, uint4 hi) {
  v8u t;
  t[0]=lo.x; t[1]=lo.y; t[2]=lo.z; t[3]=lo.w;
  t[4]=hi.x; t[5]=hi.y; t[6]=hi.z; t[7]=hi.w;
  return __builtin_bit_cast(v16bf, t);
}

// loads the two 16B K-chunks of a 16-bit WMMA operand: elems [0..7] and [16..23] from p
__device__ __forceinline__ v16bf ld_frag(const unsigned short* p) {
  uint4 lo = *reinterpret_cast<const uint4*>(p);
  uint4 hi = *reinterpret_cast<const uint4*>(p + 16);
  return mk_frag(lo, hi);
}

__device__ __forceinline__ v8f wmma_bf16(v16bf a, v16bf b, v8f c) {
  return __builtin_amdgcn_wmma_f32_16x16x32_bf16(false, a, false, b, (short)0, c, false, false);
}

// ---------------- fp32 -> bf16 conversion (vectorized) ----------------
__global__ void cvt_bf16_kernel(const float* __restrict__ s, unsigned short* __restrict__ d, int n4) {
  for (int i = blockIdx.x * blockDim.x + threadIdx.x; i < n4; i += gridDim.x * blockDim.x) {
    float4 v = reinterpret_cast<const float4*>(s)[i];
    ushort4 t;
    t.x = f2bf(v.x); t.y = f2bf(v.y); t.z = f2bf(v.z); t.w = f2bf(v.w);
    reinterpret_cast<ushort4*>(d)[i] = t;
  }
}

// ---------------- projection GEMM: out[m,n] = sum_k A[m,k]*W[n,k]  (A,W bf16) ----------------
// MODE 0: -> Qh bf16 [B,H,N,D], scaled by CF_SCALE
// MODE 1: -> Kh bf16 [B,H,N,D]
// MODE 2: -> Vt bf16 [B,H,D,N]  (transposed for PV B-operand)
// MODE 3: -> fp32 out [M,768] + bias
template <int MODE>
__global__ __launch_bounds__(256) void proj_kernel(const unsigned short* __restrict__ Abf,
                                                   const unsigned short* __restrict__ Wbf,
                                                   void* __restrict__ Out,
                                                   const float* __restrict__ bias) {
  __shared__ __align__(16) unsigned short a_lds[2][128 * 40];  // 80B row stride: conflict-free
  const int tid  = threadIdx.x;
  const int wave = tid >> 5;
  const int lane = tid & 31;
  const int col  = lane & 15;
  const int hf   = lane >> 4;
  const int n0 = blockIdx.x * 64;
  const int m0 = blockIdx.y * 128;
  const int mw = (wave & 3) * 32;
  const int nw = (wave >> 2) * 32;

  // stage 128x32 bf16 A tile (pure 16B copies -> async path)
  auto stageA = [&](int k0, int buf) {
#pragma unroll
    for (int i = tid; i < 512; i += 256) {
      const int row = i >> 2, c4 = i & 3;
      async_cp16(&a_lds[buf][row * 40 + c4 * 8],
                 Abf + (size_t)(m0 + row) * DIM + k0 + c4 * 8);
    }
  };

  v8f acc[2][2] = {};
  stageA(0, 0);

  for (int kt = 0; kt < DIM / 32; kt++) {
    const int buf = kt & 1;
    const int k0  = kt * 32;
    wait_async();
    __syncthreads();
    if (kt + 1 < DIM / 32) stageA(k0 + 32, buf ^ 1);

    v16bf af[2], bfr[2];
#pragma unroll
    for (int i = 0; i < 2; i++)
      af[i] = ld_frag(&a_lds[buf][(mw + i * 16 + col) * 40 + hf * 8]);
#pragma unroll
    for (int j = 0; j < 2; j++)
      bfr[j] = ld_frag(Wbf + (size_t)(n0 + nw + j * 16 + col) * DIM + k0 + hf * 8);
#pragma unroll
    for (int i = 0; i < 2; i++)
#pragma unroll
      for (int j = 0; j < 2; j++)
        acc[i][j] = wmma_bf16(af[i], bfr[j], acc[i][j]);
  }

  // epilogue: D elem (row = r + 8*hf, col = lane%16)
#pragma unroll
  for (int i = 0; i < 2; i++) {
#pragma unroll
    for (int j = 0; j < 2; j++) {
#pragma unroll
      for (int r = 0; r < 8; r++) {
        const int grow = m0 + mw + i * 16 + r + 8 * hf;
        const int gcol = n0 + nw + j * 16 + col;
        float v = acc[i][j][r];
        if constexpr (MODE == 3) {
          ((float*)Out)[(size_t)grow * DIM + gcol] = v + bias[gcol];
        } else {
          const int bb = grow >> 11, rr = grow & (SEQ - 1);
          const int hh = gcol / HDIM, dd = gcol % HDIM;
          unsigned short* dst = (unsigned short*)Out;
          if constexpr (MODE == 0)
            dst[(((size_t)bb * NHEADS + hh) * SEQ + rr) * HDIM + dd] = f2bf(v * CF_SCALE);
          else if constexpr (MODE == 1)
            dst[(((size_t)bb * NHEADS + hh) * SEQ + rr) * HDIM + dd] = f2bf(v);
          else
            dst[(((size_t)bb * NHEADS + hh) * HDIM + dd) * SEQ + rr] = f2bf(v);
        }
      }
    }
  }
}

// ---------------- flash attention: 8 waves x 16 Q rows, 64-key tiles, double-buffered ----------------
__global__ __launch_bounds__(256) void attn_kernel(const unsigned short* __restrict__ Qh,
                                                   const unsigned short* __restrict__ Kh,
                                                   const unsigned short* __restrict__ Vt,
                                                   unsigned short* __restrict__ AO) {
  __shared__ __align__(16) unsigned short k_lds[2][64 * 216];    // [key][d], pad->432B stride
  __shared__ __align__(16) unsigned short v_lds[2][192 * 72];    // [d][key], pad->144B stride
  __shared__ __align__(16) unsigned short p_lds[8 * 16 * 72];    // per-wave P transpose scratch

  const int tid  = threadIdx.x;
  const int wave = tid >> 5;
  const int lane = tid & 31;
  const int col  = lane & 15;
  const int hf   = lane >> 4;
  const int b = blockIdx.z, h = blockIdx.y;
  const int q0 = blockIdx.x * 128;
  const size_t bh = (size_t)(b * NHEADS + h);
  const unsigned short* Qp = Qh + bh * SEQ * HDIM;
  const unsigned short* Kp = Kh + bh * SEQ * HDIM;
  const unsigned short* Vp = Vt + bh * (size_t)HDIM * SEQ;

  auto stageKV = [&](int k0, int buf) {
    const unsigned short* ksrc = Kp + (size_t)k0 * HDIM;  // 64x192 bf16, contiguous
    for (int i = tid; i < 64 * 24; i += 256)
      async_cp16(&k_lds[buf][(i / 24) * 216 + (i % 24) * 8], ksrc + i * 8);
    for (int i = tid; i < 192 * 8; i += 256)
      async_cp16(&v_lds[buf][(i / 8) * 72 + (i % 8) * 8],
                 Vp + (size_t)(i / 8) * SEQ + k0 + (i % 8) * 8);
  };

  // Q fragments for this wave's 16 rows (scale already folded in)
  v16bf qfrag[6];
  {
    const unsigned short* qrow = Qp + (size_t)(q0 + wave * 16 + col) * HDIM;
#pragma unroll
    for (int kc = 0; kc < 6; kc++)
      qfrag[kc] = ld_frag(qrow + kc * 32 + hf * 8);
  }

  const v8f vzero = {0, 0, 0, 0, 0, 0, 0, 0};
  v8f o[12];
#pragma unroll
  for (int i = 0; i < 12; i++) o[i] = vzero;
  float mrow[8], lrow[8];
#pragma unroll
  for (int r = 0; r < 8; r++) { mrow[r] = -3.0e38f; lrow[r] = 0.0f; }

  unsigned short* pw = &p_lds[wave * 16 * 72];

  stageKV(0, 0);

  for (int t = 0; t < SEQ / 64; t++) {
    const int buf = t & 1;
    wait_async();
    __syncthreads();
    if (t + 1 < SEQ / 64) stageKV((t + 1) * 64, buf ^ 1);

    // S = Q @ K^T : 4 key-subtiles x 6 d-chunks
    v8f s[4];
#pragma unroll
    for (int n = 0; n < 4; n++) {
      v8f a = vzero;
#pragma unroll
      for (int kc = 0; kc < 6; kc++) {
        v16bf bf = ld_frag(&k_lds[buf][(n * 16 + col) * 216 + kc * 32 + hf * 8]);
        a = wmma_bf16(qfrag[kc], bf, a);
      }
      s[n] = a;
    }

    // online softmax (rows r+8*hf live in this lane half; xor<=8 stays in half)
    float mnew[8], fac[8];
#pragma unroll
    for (int r = 0; r < 8; r++) {
      float v = fmaxf(fmaxf(s[0][r], s[1][r]), fmaxf(s[2][r], s[3][r]));
#pragma unroll
      for (int off = 8; off >= 1; off >>= 1) v = fmaxf(v, __shfl_xor(v, off, 32));
      mnew[r] = fmaxf(mrow[r], v);
      fac[r]  = __builtin_amdgcn_exp2f(mrow[r] - mnew[r]);
      mrow[r] = mnew[r];
    }

    float rsum[8];
#pragma unroll
    for (int r = 0; r < 8; r++) rsum[r] = 0.0f;
#pragma unroll
    for (int n = 0; n < 4; n++) {
#pragma unroll
      for (int r = 0; r < 8; r++) {
        float p = __builtin_amdgcn_exp2f(s[n][r] - mnew[r]);
        rsum[r] += p;
        pw[(r + 8 * hf) * 72 + n * 16 + col] = f2bf(p);  // C-layout -> A-layout via LDS
      }
    }
#pragma unroll
    for (int r = 0; r < 8; r++) {
      float v = rsum[r];
#pragma unroll
      for (int off = 8; off >= 1; off >>= 1) v += __shfl_xor(v, off, 32);
      lrow[r] = lrow[r] * fac[r] + v;
    }
#pragma unroll
    for (int nf = 0; nf < 12; nf++)
#pragma unroll
      for (int r = 0; r < 8; r++) o[nf][r] *= fac[r];

    // O += P @ V : A-frags from p_lds, B-frags from Vt tile
    v16bf pf[2];
#pragma unroll
    for (int kc = 0; kc < 2; kc++)
      pf[kc] = ld_frag(&pw[col * 72 + kc * 32 + hf * 8]);
#pragma unroll
    for (int nf = 0; nf < 12; nf++) {
#pragma unroll
      for (int kc = 0; kc < 2; kc++) {
        v16bf bf = ld_frag(&v_lds[buf][(nf * 16 + col) * 72 + kc * 32 + hf * 8]);
        o[nf] = wmma_bf16(pf[kc], bf, o[nf]);
      }
    }
  }

  // epilogue: O / l, write merged-head bf16 [B*N, 768]
  float inv[8];
#pragma unroll
  for (int r = 0; r < 8; r++) inv[r] = 1.0f / lrow[r];
  const int qb = q0 + wave * 16;
#pragma unroll
  for (int nf = 0; nf < 12; nf++) {
#pragma unroll
    for (int r = 0; r < 8; r++) {
      const int q = qb + r + 8 * hf;
      AO[((size_t)(b * SEQ + q)) * DIM + h * HDIM + nf * 16 + col] = f2bf(o[nf][r] * inv[r]);
    }
  }
}

extern "C" void kernel_launch(void* const* d_in, const int* in_sizes, int n_in,
                              void* d_out, int out_size, void* d_ws, size_t ws_size,
                              hipStream_t stream) {
  const float* q  = (const float*)d_in[0];
  const float* k  = (const float*)d_in[1];
  const float* v  = (const float*)d_in[2];
  const float* Wq = (const float*)d_in[3];
  const float* Wk = (const float*)d_in[4];
  const float* Wv = (const float*)d_in[5];
  const float* Wp = (const float*)d_in[6];
  const float* bp = (const float*)d_in[7];
  float* out = (float*)d_out;

  unsigned short* ws = (unsigned short*)d_ws;
  unsigned short* wq_bf = ws;
  unsigned short* wk_bf = wq_bf + (size_t)WELEMS;
  unsigned short* wv_bf = wk_bf + (size_t)WELEMS;
  unsigned short* wp_bf = wv_bf + (size_t)WELEMS;
  unsigned short* q_bf  = wp_bf + (size_t)WELEMS;
  unsigned short* k_bf  = q_bf + AELEMS;
  unsigned short* v_bf  = k_bf + AELEMS;
  unsigned short* Qh    = v_bf + AELEMS;
  unsigned short* Kh    = Qh + HELEMS;
  unsigned short* Vt    = Kh + HELEMS;
  unsigned short* AO    = Vt + HELEMS;

  cvt_bf16_kernel<<<576,  256, 0, stream>>>(Wq, wq_bf, WELEMS / 4);
  cvt_bf16_kernel<<<576,  256, 0, stream>>>(Wk, wk_bf, WELEMS / 4);
  cvt_bf16_kernel<<<576,  256, 0, stream>>>(Wv, wv_bf, WELEMS / 4);
  cvt_bf16_kernel<<<576,  256, 0, stream>>>(Wp, wp_bf, WELEMS / 4);
  cvt_bf16_kernel<<<4096, 256, 0, stream>>>(q, q_bf, (int)(AELEMS / 4));
  cvt_bf16_kernel<<<4096, 256, 0, stream>>>(k, k_bf, (int)(AELEMS / 4));
  cvt_bf16_kernel<<<4096, 256, 0, stream>>>(v, v_bf, (int)(AELEMS / 4));

  dim3 pgrid(DIM / 64, MROWS / 128);  // 12 x 128
  proj_kernel<0><<<pgrid, 256, 0, stream>>>(q_bf, wq_bf, Qh, nullptr);
  proj_kernel<1><<<pgrid, 256, 0, stream>>>(k_bf, wk_bf, Kh, nullptr);
  proj_kernel<2><<<pgrid, 256, 0, stream>>>(v_bf, wv_bf, Vt, nullptr);

  attn_kernel<<<dim3(SEQ / 128, NHEADS, BATCH), 256, 0, stream>>>(Qh, Kh, Vt, AO);

  proj_kernel<3><<<pgrid, 256, 0, stream>>>(AO, wp_bf, out, bp);
}
